// CausalSelfAttention_63024350101680
// MI455X (gfx1250) — matrix operand order repaired and checked by reading, not compiled, not run on hardware
//
#include <hip/hip_runtime.h>
#include <hip/hip_bf16.h>

typedef __attribute__((ext_vector_type(4)))  _Float16 v4h;
typedef __attribute__((ext_vector_type(8)))  _Float16 v8h;
typedef __attribute__((ext_vector_type(16))) _Float16 v16h;
typedef __attribute__((ext_vector_type(8)))  float    v8f;

// Problem constants: B=4, S=2048, D=1024, H=16, HD=64
#define CB  4
#define CS  2048
#define CD  1024
#define CH  16
#define CHD 64
#define CM  (CB * CS)                             // 8192 rows
#define HEAD_ELEMS ((size_t)CB * CH * CS * CHD)   // 8388608 per q/k/v buffer

__device__ __forceinline__ v16h cat16(v8h lo, v8h hi) {
    return __builtin_shufflevector(lo, hi, 0,1,2,3,4,5,6,7,8,9,10,11,12,13,14,15);
}
__device__ __forceinline__ v8f wmma_f16(v16h a, v16h b, v8f c) {
    return __builtin_amdgcn_wmma_f32_16x16x32_f16(
        false, a, false, b, (short)0, c, false, false);
}
__device__ __forceinline__ v8h ld8(const _Float16* p) { return *(const v8h*)p; }

// ---------------------------------------------------------------------------
// fp32 -> fp16 elementwise convert (vector-4)
// ---------------------------------------------------------------------------
__global__ __launch_bounds__(256) void cvt_f16(
    const float* __restrict__ in, _Float16* __restrict__ out, int n)
{
    const int i = (blockIdx.x * 256 + threadIdx.x) * 4;
    if (i < n) {
        float4 v = *(const float4*)(in + i);
        v4h h = { (_Float16)v.x, (_Float16)v.y, (_Float16)v.z, (_Float16)v.w };
        *(v4h*)(out + i) = h;
    }
}

// ---------------------------------------------------------------------------
// W[K][N] fp32 -> Wt[N][K] fp16, tiled 32x32 through LDS
// ---------------------------------------------------------------------------
__global__ __launch_bounds__(256) void transpose_cvt(
    const float* __restrict__ W, _Float16* __restrict__ Wt, int K, int N)
{
    __shared__ float tile[32][33];
    const int k0 = blockIdx.y * 32, n0 = blockIdx.x * 32;
    const int tx = threadIdx.x & 31, ty = threadIdx.x >> 5;  // 32 x 8
    #pragma unroll
    for (int r = ty; r < 32; r += 8)
        tile[r][tx] = W[(size_t)(k0 + r) * N + n0 + tx];
    __syncthreads();
    #pragma unroll
    for (int r = ty; r < 32; r += 8)
        Wt[(size_t)(n0 + r) * K + k0 + tx] = (_Float16)tile[tx][r];
}

// ---------------------------------------------------------------------------
// f16 WMMA GEMM:  C[M,N] = A[M,K] @ B[K,N] + bias[N],  B given transposed Bt[N][K].
// Block = 8 waves (4M x 2N), wave tile = 32x64 (2x4 accumulators) => block 128x128.
// Fragment layouts used here (16-bit, 16x16x32; contiguous k, same order in A and B):
//   A: lane half 0 holds K = [k..k+15], half 1 K = [k+16..k+31] (contig b128 x2)
//   B: lane half 0 holds K = [k..k+15], half 1 K = [k+16..k+31] (contig from Bt row)
//   C (f32): VGPR r -> (M=r, N=lane) lanes 0-15, (M=r+8, N=lane-16) lanes 16-31
// MODE 0: fp32 row-major store.  MODE 1: qkv scatter to f16 q[BHS,HD]/k[BHS,HD]/vT[BH,HD,S].
// ---------------------------------------------------------------------------
template <int MODE>
__global__ __launch_bounds__(256) void hgemm_wmma(
    const _Float16* __restrict__ Ah,   // [M][K]
    const _Float16* __restrict__ Bt,   // [N][K]
    const float* __restrict__ bias,    // [N]
    void* __restrict__ outp,
    int M, int N, int K)
{
    const int lane = threadIdx.x & 31;
    const int wave = threadIdx.x >> 5;
    const int wm = wave >> 1, wn = wave & 1;
    const int half = lane >> 4, l16 = lane & 15;

    const int tileM = blockIdx.y * 128 + wm * 32;
    const int tileN = blockIdx.x * 128 + wn * 64;

    const _Float16* __restrict__ arow0 = Ah + (size_t)(tileM + l16) * K;
    const _Float16* __restrict__ arow1 = arow0 + (size_t)16 * K;
    const _Float16* __restrict__ brow0 = Bt + (size_t)(tileN + 0 * 16 + l16) * K;
    const _Float16* __restrict__ brow1 = Bt + (size_t)(tileN + 1 * 16 + l16) * K;
    const _Float16* __restrict__ brow2 = Bt + (size_t)(tileN + 2 * 16 + l16) * K;
    const _Float16* __restrict__ brow3 = Bt + (size_t)(tileN + 3 * 16 + l16) * K;

    v8f acc[2][4] = {};

    #pragma unroll 2
    for (int k = 0; k < K; k += 32) {
        const int ao = k + half * 16;
        const int bo = k + half * 16;
        __builtin_prefetch(arow0 + ao + 256, 0, 3);   // WGP-scope near prefetch
        v16h a0 = cat16(ld8(arow0 + ao), ld8(arow0 + ao + 8));
        v16h a1 = cat16(ld8(arow1 + ao), ld8(arow1 + ao + 8));
        v16h b0 = cat16(ld8(brow0 + bo), ld8(brow0 + bo + 8));
        v16h b1 = cat16(ld8(brow1 + bo), ld8(brow1 + bo + 8));
        v16h b2 = cat16(ld8(brow2 + bo), ld8(brow2 + bo + 8));
        v16h b3 = cat16(ld8(brow3 + bo), ld8(brow3 + bo + 8));
        acc[0][0] = wmma_f16(a0, b0, acc[0][0]);
        acc[1][0] = wmma_f16(a1, b0, acc[1][0]);
        acc[0][1] = wmma_f16(a0, b1, acc[0][1]);
        acc[1][1] = wmma_f16(a1, b1, acc[1][1]);
        acc[0][2] = wmma_f16(a0, b2, acc[0][2]);
        acc[1][2] = wmma_f16(a1, b2, acc[1][2]);
        acc[0][3] = wmma_f16(a0, b3, acc[0][3]);
        acc[1][3] = wmma_f16(a1, b3, acc[1][3]);
    }

    #pragma unroll
    for (int mi = 0; mi < 2; ++mi) {
        #pragma unroll
        for (int ni = 0; ni < 4; ++ni) {
            const int n = tileN + ni * 16 + l16;
            const float bv = bias[n];
            #pragma unroll
            for (int r = 0; r < 8; ++r) {
                const int m = tileM + mi * 16 + r + half * 8;
                const float v = acc[mi][ni][r] + bv;
                if constexpr (MODE == 0) {
                    ((float*)outp)[(size_t)m * N + n] = v;
                } else {
                    // scatter into head-major f16 q/k and transposed v
                    const int bb = m >> 11;              // / CS
                    const int s  = m & (CS - 1);
                    const int sect = n >> 10;            // 0=q 1=k 2=v
                    const int h    = (n & (CD - 1)) >> 6;
                    const int d    = n & (CHD - 1);
                    _Float16* base = (_Float16*)outp;
                    const _Float16 hv = (_Float16)v;
                    const size_t head = (size_t)(bb * CH + h);
                    if (sect == 0)
                        base[(head * CS + s) * CHD + d] = hv;
                    else if (sect == 1)
                        base[HEAD_ELEMS + (head * CS + s) * CHD + d] = hv;
                    else
                        base[2 * HEAD_ELEMS + (head * CHD + d) * CS + s] = hv;
                }
            }
        }
    }
}

// ---------------------------------------------------------------------------
// f16 WMMA flash attention (causal). One block = 8 waves = 128 query rows of
// one (b,h); each wave privately owns 16 rows, iterates 32-key blocks:
//   scores (2 tiles x 2 K-chunks = 4 WMMAs) -> fp32 online softmax via
//   wave-private LDS -> P(f16) @ V (4 WMMAs over HD=64).
// All fragments are contiguous b128 pairs thanks to q/k [B,H,S,HD] and
// vT [B,H,HD,S] layouts. K and V fragments are loaded into distinct registers
// before the WMMA chains so loads stay in flight under the matrix ops.
// ---------------------------------------------------------------------------
__global__ __launch_bounds__(256) void attn_wmma_f16(
    const _Float16* __restrict__ qh,
    const _Float16* __restrict__ kh,
    const _Float16* __restrict__ vth,
    _Float16* __restrict__ yh)        // [B, S, D] f16
{
    constexpr int NQB = CS / 128;
    const float att_scale = 0.125f;   // 1/sqrt(HD)

    __shared__ float lds_s[8][16][33];                      // fp32 scores
    __shared__ __align__(16) _Float16 lds_p[8][16][40];     // f16 probabilities
    __shared__ float lds_alpha[8][16];
    __shared__ float lds_l[8][16];

    const int lane = threadIdx.x & 31;
    const int wave = threadIdx.x >> 5;
    const int half = lane >> 4, l16 = lane & 15;

    const int bh    = blockIdx.x / NQB;
    const int bi    = bh / CH;
    const int h     = bh % CH;
    const int qbase = (blockIdx.x % NQB) * 128 + wave * 16;

    const size_t head = (size_t)(bi * CH + h);
    const _Float16* __restrict__ qp = qh  + head * CS * CHD;
    const _Float16* __restrict__ kp = kh  + head * CS * CHD;
    const _Float16* __restrict__ vp = vth + head * CHD * CS;

    // Preload Q tile (16 rows x 64) as two A-fragments per lane.
    v16h qa0, qa1;
    {
        const _Float16* qrow = qp + (size_t)(qbase + l16) * CHD;
        qa0 = cat16(ld8(qrow + half * 16),      ld8(qrow + half * 16 + 8));
        qa1 = cat16(ld8(qrow + 32 + half * 16), ld8(qrow + 32 + half * 16 + 8));
    }

    v8f o0 = {}, o1 = {}, o2 = {}, o3 = {};
    float m_run = -1.0e30f, l_run = 0.0f;   // valid on lanes 0-15 (lane == row)

    const int nKB = qbase / 32 + 1;          // causal
    for (int kb = 0; kb < nKB; ++kb) {
        const int kbase = kb * 32;

        // ---- scores = Q @ K^T : two 16x16 tiles, K-dim = HD = 64 ----
        const _Float16* krow0 = kp + (size_t)(kbase + l16) * CHD;
        const _Float16* krow1 = krow0 + 16 * CHD;
        __builtin_prefetch(krow0 + 32 * CHD, 0, 3);             // next K block
        __builtin_prefetch(vp + (size_t)l16 * CS + kbase, 0, 3);// V rows, tile 0

        // load all four K fragments first, then issue the WMMA chain
        v16h bk00 = cat16(ld8(krow0 + half * 16),      ld8(krow0 + half * 16 + 8));
        v16h bk01 = cat16(ld8(krow0 + 32 + half * 16), ld8(krow0 + 32 + half * 16 + 8));
        v16h bk10 = cat16(ld8(krow1 + half * 16),      ld8(krow1 + half * 16 + 8));
        v16h bk11 = cat16(ld8(krow1 + 32 + half * 16), ld8(krow1 + 32 + half * 16 + 8));
        v8f c0 = {}, c1 = {};
        c0 = wmma_f16(qa0, bk00, c0);
        c1 = wmma_f16(qa0, bk10, c1);
        c0 = wmma_f16(qa1, bk01, c0);
        c1 = wmma_f16(qa1, bk11, c1);

        // spill scaled scores to wave-private LDS [16 rows][32 keys]
        #pragma unroll
        for (int r = 0; r < 8; ++r) {
            lds_s[wave][r + half * 8][l16]      = c0[r] * att_scale;
            lds_s[wave][r + half * 8][16 + l16] = c1[r] * att_scale;
        }

        // ---- online softmax: lane q (0..15) owns query row qbase+q ----
        if (lane < 16) {
            const int qrow = qbase + l16;
            float s[32];
            float mloc = -3.0e38f;
            #pragma unroll
            for (int j = 0; j < 32; ++j) {
                float v = lds_s[wave][l16][j];
                if (kbase + j > qrow) v = -3.0e38f;     // causal mask
                s[j] = v;
                mloc = fmaxf(mloc, v);
            }
            const float mnew  = fmaxf(m_run, mloc);
            const float alpha = __expf(m_run - mnew);
            float lsum = 0.0f;
            #pragma unroll
            for (int j = 0; j < 32; ++j) {
                float p = __expf(s[j] - mnew);
                if (kbase + j > qrow) p = 0.0f;
                lds_p[wave][l16][j] = (_Float16)p;
                lsum += p;
            }
            l_run = l_run * alpha + lsum;
            m_run = mnew;
            lds_alpha[wave][l16] = alpha;
        }

        // ---- load V fragments (4 col tiles) while alpha is being read ----
        const _Float16* vr0 = vp + (size_t)( 0 + l16) * CS + kbase + half * 16;
        const _Float16* vr1 = vp + (size_t)(16 + l16) * CS + kbase + half * 16;
        const _Float16* vr2 = vp + (size_t)(32 + l16) * CS + kbase + half * 16;
        const _Float16* vr3 = vp + (size_t)(48 + l16) * CS + kbase + half * 16;
        v16h vb0 = cat16(ld8(vr0), ld8(vr0 + 8));
        v16h vb1 = cat16(ld8(vr1), ld8(vr1 + 8));
        v16h vb2 = cat16(ld8(vr2), ld8(vr2 + 8));
        v16h vb3 = cat16(ld8(vr3), ld8(vr3 + 8));

        // ---- rescale accumulators by per-row alpha ----
        float af[8];
        #pragma unroll
        for (int r = 0; r < 8; ++r) af[r] = lds_alpha[wave][r + half * 8];
        #pragma unroll
        for (int r = 0; r < 8; ++r) {
            o0[r] *= af[r]; o1[r] *= af[r]; o2[r] *= af[r]; o3[r] *= af[r];
        }

        // ---- O += P @ V : P A-frag from LDS, V B-frags from registers ----
        const _Float16* prow = &lds_p[wave][l16][0];
        v16h pa = cat16(ld8(prow + half * 16), ld8(prow + half * 16 + 8));
        o0 = wmma_f16(pa, vb0, o0);
        o1 = wmma_f16(pa, vb1, o1);
        o2 = wmma_f16(pa, vb2, o2);
        o3 = wmma_f16(pa, vb3, o3);
    }

    // ---- final 1/l normalization, store f16 y[B,S,D] ----
    if (lane < 16) lds_l[wave][l16] = l_run;
    float lf[8];
    #pragma unroll
    for (int r = 0; r < 8; ++r) lf[r] = lds_l[wave][r + half * 8];

    _Float16* __restrict__ ybase = yh + (size_t)bi * CS * CD + h * CHD;
    #pragma unroll
    for (int r = 0; r < 8; ++r) {
        const int row = qbase + r + half * 8;
        const float inv = 1.0f / lf[r];
        _Float16* yr = ybase + (size_t)row * CD;
        yr[ 0 + l16] = (_Float16)(o0[r] * inv);
        yr[16 + l16] = (_Float16)(o1[r] * inv);
        yr[32 + l16] = (_Float16)(o2[r] * inv);
        yr[48 + l16] = (_Float16)(o3[r] * inv);
    }
}

// ---------------------------------------------------------------------------
// Launch: convert/transpose -> QKV GEMM (scatter) -> attention -> out GEMM.
// ---------------------------------------------------------------------------
extern "C" void kernel_launch(void* const* d_in, const int* in_sizes, int n_in,
                              void* d_out, int out_size, void* d_ws, size_t ws_size,
                              hipStream_t stream) {
    const float* x     = (const float*)d_in[0];   // [B,S,D]
    const float* w_qkv = (const float*)d_in[1];   // [D,3D]
    const float* b_qkv = (const float*)d_in[2];   // [3D]
    const float* w_out = (const float*)d_in[3];   // [D,D]
    const float* b_out = (const float*)d_in[4];   // [D]
    float* out = (float*)d_out;                   // [B,S,D] fp32

    // Workspace carve-up (all f16)
    _Float16* p   = (_Float16*)d_ws;
    _Float16* xh  = p;  p += (size_t)CM * CD;          //  x   [8192][1024]
    _Float16* wqh = p;  p += (size_t)3 * CD * CD;      //  wqkv^T [3072][1024]
    _Float16* woh = p;  p += (size_t)CD * CD;          //  wout^T [1024][1024]
    _Float16* qh  = p;  p += HEAD_ELEMS;               //  q  [B,H,S,HD]
    /* kh, vth are contiguous after qh (scatter epilogue indexes them) */
    _Float16* kh  = p;  p += HEAD_ELEMS;               //  k  [B,H,S,HD]
    _Float16* vth = p;  p += HEAD_ELEMS;               //  vT [B,H,HD,S]
    _Float16* yh  = p;  p += (size_t)CM * CD;          //  y  [8192][1024]

    // 0) conversions
    cvt_f16<<<dim3((CM * CD) / (256 * 4)), 256, 0, stream>>>(x, xh, CM * CD);
    transpose_cvt<<<dim3((3 * CD) / 32, CD / 32), 256, 0, stream>>>(w_qkv, wqh, CD, 3 * CD);
    transpose_cvt<<<dim3(CD / 32, CD / 32), 256, 0, stream>>>(w_out, woh, CD, CD);

    // 1) qkv = x @ w_qkv + b_qkv, scattered into q/k/vT head layouts
    hgemm_wmma<1><<<dim3((3 * CD) / 128, CM / 128), 256, 0, stream>>>(
        xh, wqh, b_qkv, (void*)qh, CM, 3 * CD, CD);

    // 2) causal flash attention
    attn_wmma_f16<<<dim3(CB * CH * (CS / 128)), 256, 0, stream>>>(qh, kh, vth, yh);

    // 3) out = y @ w_out + b_out (fp32 store)
    hgemm_wmma<0><<<dim3(CD / 128, CM / 128), 256, 0, stream>>>(
        yh, woh, b_out, (void*)out, CM, CD, CD);
}
